// GATv2_8847632630483
// MI455X (gfx1250) — compile-verified, matrix operand order checked by reading, not run on hardware
//
#include <hip/hip_runtime.h>
#include <cstdint>
#include <cstddef>

#define NNODES 50000
#define NEDGES 800000
#define NHEADS 4

typedef _Float16 v8h  __attribute__((ext_vector_type(8)));
typedef _Float16 v16h __attribute__((ext_vector_type(16)));
typedef float    v8f  __attribute__((ext_vector_type(8)));

// ===========================================================================
// Projection: f32 -> f16 operand prep
// ===========================================================================
__global__ void cvt_h_f16(const float* __restrict__ h, _Float16* __restrict__ h16,
                          int n, int fin, int kpad) {
  int i = blockIdx.x * blockDim.x + threadIdx.x;
  if (i >= n * kpad) return;
  int r = i / kpad, k = i - r * kpad;
  h16[i] = (k < fin) ? (_Float16)h[r * fin + k] : (_Float16)0.0f;
}

// W [K, fout] f32 -> w16t [fout, kpad] f16 (transposed, zero-padded K)
__global__ void cvt_w_f16t(const float* __restrict__ W, _Float16* __restrict__ w16t,
                           int K, int fout, int kpad) {
  int i = blockIdx.x * blockDim.x + threadIdx.x;
  if (i >= fout * kpad) return;
  int o = i / kpad, k = i - o * kpad;
  w16t[i] = (k < K) ? (_Float16)W[k * fout + o] : (_Float16)0.0f;
}

// ===========================================================================
// WMMA GEMM: C[n, FOUT] = A16[n, KPAD] * BT[FOUT, KPAD]^T
// One 128-thread block (4 waves) per 16-row stripe. The weight matrix
// (FOUT*KPAD*2 bytes <= 32KB) is staged into LDS via the CDNA5 async copy
// path (global_load_async_to_lds_b128 / s_wait_asynccnt), then each wave
// computes 16x16 tiles with v_wmma_f32_16x16x32_f16, reusing the A row
// fragments (registers) across all column tiles.
// ===========================================================================
template <int KPAD, int FOUT>
__global__ void __launch_bounds__(128)
gemm_wmma(const _Float16* __restrict__ A, const _Float16* __restrict__ BTg,
          float* __restrict__ C) {
  __shared__ alignas(16) _Float16 bs[FOUT * KPAD];
  const int tid = threadIdx.x;

  // ---- async stage BT into LDS: 16B chunks, consecutive lanes -> contiguous
  constexpr int CHUNKS = (FOUT * KPAD) / 8;  // 8 halfs = 16B per chunk
#pragma unroll
  for (int i = tid; i < CHUNKS; i += 128) {
    unsigned loff = (unsigned)(uintptr_t)&bs[i << 3];
    const _Float16* g = BTg + ((size_t)i << 3);
    asm volatile("global_load_async_to_lds_b128 %0, %1, off"
                 :: "v"(loff), "v"(g) : "memory");
  }
  asm volatile("s_wait_asynccnt 0" ::: "memory");
  __syncthreads();

  const int lane = tid & 31;
  const int wid  = tid >> 5;
  const int m    = lane & 15;
  const int kh   = (lane >> 4) << 3;  // 0 or 8: K-halves per CDNA5 16-bit layout

  // ---- A fragments for this 16-row stripe, loaded once, reused per col tile
  constexpr int NKF = KPAD / 32;
  v16h afrag[NKF];
  {
    const _Float16* ap = A + (size_t)(blockIdx.x * 16 + m) * KPAD + kh;
#pragma unroll
    for (int kf = 0; kf < NKF; kf++) {
      v8h lo = *(const v8h*)(ap + kf * 32);
      v8h hi = *(const v8h*)(ap + kf * 32 + 16);
#pragma unroll
      for (int i = 0; i < 8; i++) { afrag[kf][i] = lo[i]; afrag[kf][i + 8] = hi[i]; }
    }
  }

  constexpr int NTILES = FOUT / 16;
  for (int ct = wid; ct < NTILES; ct += 4) {
    v8f acc = {};
    const _Float16* bp = &bs[(size_t)(ct * 16 + m) * KPAD + kh];
#pragma unroll
    for (int kf = 0; kf < NKF; kf++) {
      v8h lo = *(const v8h*)(bp + kf * 32);   // ds_load_b128
      v8h hi = *(const v8h*)(bp + kf * 32 + 16);
      v16h b;
#pragma unroll
      for (int i = 0; i < 8; i++) { b[i] = lo[i]; b[i + 8] = hi[i]; }
      acc = __builtin_amdgcn_wmma_f32_16x16x32_f16(
          false, afrag[kf], false, b, (short)0, acc, false, false);
    }
    // D layout: VGPR i -> row i (lanes 0-15) / i+8 (lanes 16-31), col = lane&15
    int rbase = blockIdx.x * 16 + ((lane >> 4) << 3);
    int col   = ct * 16 + m;
#pragma unroll
    for (int i = 0; i < 8; i++)
      C[(size_t)(rbase + i) * FOUT + col] = acc[i];
  }
}

// ===========================================================================
// CSR build (dst is constant across all 4 layers -> built once per launch)
// ===========================================================================
__global__ void k_zero_int(int* __restrict__ p, int n) {
  int i = blockIdx.x * blockDim.x + threadIdx.x;
  if (i < n) p[i] = 0;
}

__global__ void k_hist(const int* __restrict__ dst, int* __restrict__ deg) {
  int e = blockIdx.x * blockDim.x + threadIdx.x;
  if (e < NEDGES) atomicAdd(&deg[dst[e]], 1);
}

__global__ void k_bsum(const int* __restrict__ deg, int* __restrict__ bsum, int n) {
  __shared__ int s[256];
  int i = blockIdx.x * 256 + threadIdx.x;
  s[threadIdx.x] = (i < n) ? deg[i] : 0;
  __syncthreads();
#pragma unroll
  for (int st = 128; st; st >>= 1) {
    if (threadIdx.x < st) s[threadIdx.x] += s[threadIdx.x + st];
    __syncthreads();
  }
  if (threadIdx.x == 0) bsum[blockIdx.x] = s[0];
}

__global__ void k_bscan(int* __restrict__ bsum, int nb, int* __restrict__ rowptr_last) {
  if (blockIdx.x == 0 && threadIdx.x == 0) {
    int acc = 0;
    for (int i = 0; i < nb; i++) { int v = bsum[i]; bsum[i] = acc; acc += v; }
    rowptr_last[0] = acc;  // rowptr[N] = E
  }
}

__global__ void k_scan_final(const int* __restrict__ deg, const int* __restrict__ bsum,
                             int* __restrict__ rowptr, int* __restrict__ cursor, int n) {
  __shared__ int s[256];
  int i = blockIdx.x * 256 + threadIdx.x;
  int v = (i < n) ? deg[i] : 0;
  s[threadIdx.x] = v;
  __syncthreads();
#pragma unroll
  for (int off = 1; off < 256; off <<= 1) {  // Hillis-Steele inclusive scan
    int t = (threadIdx.x >= off) ? s[threadIdx.x - off] : 0;
    __syncthreads();
    s[threadIdx.x] += t;
    __syncthreads();
  }
  if (i < n) {
    int ex = bsum[blockIdx.x] + s[threadIdx.x] - v;  // exclusive prefix
    rowptr[i] = ex;
    cursor[i] = ex;
  }
}

__global__ void k_fill(const int* __restrict__ dst, int* __restrict__ cursor,
                       int* __restrict__ eidx) {
  int e = blockIdx.x * blockDim.x + threadIdx.x;
  if (e < NEDGES) {
    int p = atomicAdd(&cursor[dst[e]], 1);
    eidx[p] = e;
  }
}

// ===========================================================================
// one wave32 per edge: e[h] = sum_d leaky_relu(feat[s]+feat[d]) * attn
// butterfly-reduce 4 head partials; store raw scores (no atomics needed)
// ===========================================================================
template <int FOUT, int DSHIFT>
__global__ void edge_score(const float* __restrict__ feat, const int* __restrict__ src,
                           const int* __restrict__ dst, const float* __restrict__ attn,
                           float* __restrict__ esc) {
  int gid  = blockIdx.x * blockDim.x + threadIdx.x;
  int e    = gid >> 5;
  int lane = gid & 31;
  if (e >= NEDGES) return;
  int s = src[e], d = dst[e];
  const float* fs = feat + (size_t)s * FOUT;
  const float* fd = feat + (size_t)d * FOUT;
  __builtin_prefetch(fs, 0, 1);   // global_prefetch_b8
  __builtin_prefetch(fd, 0, 1);
  float hs[NHEADS] = {0.f, 0.f, 0.f, 0.f};
#pragma unroll
  for (int t = 0; t < FOUT / 32; t++) {
    int j = lane + 32 * t;
    float v = fs[j] + fd[j];
    v = (v > 0.0f) ? v : 0.2f * v;  // leaky relu
    hs[j >> DSHIFT] += v * attn[j];
  }
#pragma unroll
  for (int msk = 16; msk > 0; msk >>= 1) {
#pragma unroll
    for (int h = 0; h < NHEADS; h++) hs[h] += __shfl_xor(hs[h], msk, 32);
  }
  if (lane < NHEADS) esc[(size_t)e * NHEADS + lane] = hs[lane];
}

// ===========================================================================
// one wave32 per node: atomic-free edge-softmax + weighted aggregation.
// Pass 1/2: per-head max & sum over incident edges (lane-parallel over edges,
// butterfly reduce). Pass 3: sequential over incident edges; lanes span the
// FOUT features (coalesced 128B loads), accumulate in registers, fused ELU.
// ===========================================================================
template <int FOUT, int DSHIFT, bool ACT>
__global__ void __launch_bounds__(256)
node_aggregate(const float* __restrict__ feat, const float* __restrict__ esc,
               const int* __restrict__ rowptr, const int* __restrict__ eidx,
               const int* __restrict__ src, float* __restrict__ rst) {
  int gid  = blockIdx.x * blockDim.x + threadIdx.x;
  int n    = gid >> 5;
  int lane = gid & 31;
  if (n >= NNODES) return;
  int s0 = rowptr[n], s1 = rowptr[n + 1];

  float mx[NHEADS] = {-1e30f, -1e30f, -1e30f, -1e30f};
  for (int i = s0 + lane; i < s1; i += 32) {
    int e = eidx[i];
#pragma unroll
    for (int h = 0; h < NHEADS; h++) mx[h] = fmaxf(mx[h], esc[(size_t)e * NHEADS + h]);
  }
#pragma unroll
  for (int msk = 16; msk > 0; msk >>= 1)
#pragma unroll
    for (int h = 0; h < NHEADS; h++) mx[h] = fmaxf(mx[h], __shfl_xor(mx[h], msk, 32));

  float sm[NHEADS] = {0.f, 0.f, 0.f, 0.f};
  for (int i = s0 + lane; i < s1; i += 32) {
    int e = eidx[i];
#pragma unroll
    for (int h = 0; h < NHEADS; h++) sm[h] += __expf(esc[(size_t)e * NHEADS + h] - mx[h]);
  }
#pragma unroll
  for (int msk = 16; msk > 0; msk >>= 1)
#pragma unroll
    for (int h = 0; h < NHEADS; h++) sm[h] += __shfl_xor(sm[h], msk, 32);
#pragma unroll
  for (int h = 0; h < NHEADS; h++) sm[h] = (sm[h] > 0.f) ? 1.f / sm[h] : 0.f;

  constexpr int T = FOUT / 32;
  float acc[T];
#pragma unroll
  for (int t = 0; t < T; t++) acc[t] = 0.f;

  for (int i = s0; i < s1; i++) {
    int e = eidx[i];
    int sn = src[e];
    const float* fs = feat + (size_t)sn * FOUT;
    float ae[NHEADS];
#pragma unroll
    for (int h = 0; h < NHEADS; h++)
      ae[h] = __expf(esc[(size_t)e * NHEADS + h] - mx[h]) * sm[h];
#pragma unroll
    for (int t = 0; t < T; t++) {
      int j = lane + 32 * t;
      acc[t] += fs[j] * ae[j >> DSHIFT];
    }
  }
#pragma unroll
  for (int t = 0; t < T; t++) {
    int j = lane + 32 * t;
    float v = acc[t];
    if (ACT) v = (v > 0.f) ? v : (__expf(v) - 1.f);  // fused ELU
    rst[(size_t)n * FOUT + j] = v;
  }
}

// ===========================================================================
// final: mean over 4 heads of [N,4,8] -> [N,8]
// ===========================================================================
__global__ void head_mean(const float* __restrict__ rst, float* __restrict__ out, int n) {
  int i = blockIdx.x * blockDim.x + threadIdx.x;
  if (i >= n * 8) return;
  int r = i >> 3, c = i & 7;
  const float* p = rst + (size_t)r * 32;
  out[i] = 0.25f * (p[c] + p[8 + c] + p[16 + c] + p[24 + c]);
}

// ===========================================================================
extern "C" void kernel_launch(void* const* d_in, const int* in_sizes, int n_in,
                              void* d_out, int out_size, void* d_ws, size_t ws_size,
                              hipStream_t stream) {
  (void)in_sizes; (void)n_in; (void)out_size; (void)ws_size;
  const float* feats   = (const float*)d_in[0];
  const float* Wl[4]   = {(const float*)d_in[1], (const float*)d_in[3],
                          (const float*)d_in[5], (const float*)d_in[7]};
  const float* attn[4] = {(const float*)d_in[2], (const float*)d_in[4],
                          (const float*)d_in[6], (const float*)d_in[8]};
  const int* src = (const int*)d_in[9];
  const int* dst = (const int*)d_in[10];
  float* out = (float*)d_out;

  char* wsp = (char*)d_ws;
  auto alloc = [&](size_t bytes) -> char* {
    char* p = wsp;
    wsp += (bytes + 255) & ~(size_t)255;
    return p;
  };
  _Float16* h16  = (_Float16*)alloc((size_t)NNODES * 128 * 2);
  _Float16* w16t = (_Float16*)alloc((size_t)128 * 128 * 2);
  float* feat   = (float*)alloc((size_t)NNODES * 128 * 4);
  float* esc    = (float*)alloc((size_t)NEDGES * NHEADS * 4);
  float* bufA   = (float*)alloc((size_t)NNODES * 128 * 4);
  float* bufB   = (float*)alloc((size_t)NNODES * 128 * 4);
  int*   deg    = (int*)alloc((size_t)(NNODES + 1) * 4);
  int*   rowptr = (int*)alloc((size_t)(NNODES + 1) * 4);
  int*   cursor = (int*)alloc((size_t)NNODES * 4);
  int*   bsum   = (int*)alloc((size_t)1024 * 4);
  int*   eidx   = (int*)alloc((size_t)NEDGES * 4);

  const int NBLK = (NNODES + 255) / 256;  // 196

  // ---- CSR build (once; dst identical for all 4 layers) ----
  k_zero_int<<<NBLK, 256, 0, stream>>>(deg, NNODES);
  k_hist<<<(NEDGES + 255) / 256, 256, 0, stream>>>(dst, deg);
  k_bsum<<<NBLK, 256, 0, stream>>>(deg, bsum, NNODES);
  k_bscan<<<1, 1, 0, stream>>>(bsum, NBLK, rowptr + NNODES);
  k_scan_final<<<NBLK, 256, 0, stream>>>(deg, bsum, rowptr, cursor, NNODES);
  k_fill<<<(NEDGES + 255) / 256, 256, 0, stream>>>(dst, cursor, eidx);

  const int EG = (NEDGES * 32 + 255) / 256;
  const int NG = (NNODES * 32 + 255) / 256;

  auto proj = [&](const float* hin, int Fin, const float* W, int Fout, int kpad) {
    int t = NNODES * kpad;
    cvt_h_f16<<<(t + 255) / 256, 256, 0, stream>>>(hin, h16, NNODES, Fin, kpad);
    t = Fout * kpad;
    cvt_w_f16t<<<(t + 255) / 256, 256, 0, stream>>>(W, w16t, Fin, Fout, kpad);
  };

  // ---- layer 0: Fin=16 (kpad 32), Fout=128, ELU ----
  proj(feats, 16, Wl[0], 128, 32);
  gemm_wmma<32, 128><<<NNODES / 16, 128, 0, stream>>>(h16, w16t, feat);
  edge_score<128, 5><<<EG, 256, 0, stream>>>(feat, src, dst, attn[0], esc);
  node_aggregate<128, 5, true><<<NG, 256, 0, stream>>>(feat, esc, rowptr, eidx, src, bufA);

  // ---- layer 1 ----
  proj(bufA, 128, Wl[1], 128, 128);
  gemm_wmma<128, 128><<<NNODES / 16, 128, 0, stream>>>(h16, w16t, feat);
  edge_score<128, 5><<<EG, 256, 0, stream>>>(feat, src, dst, attn[1], esc);
  node_aggregate<128, 5, true><<<NG, 256, 0, stream>>>(feat, esc, rowptr, eidx, src, bufB);

  // ---- layer 2 ----
  proj(bufB, 128, Wl[2], 128, 128);
  gemm_wmma<128, 128><<<NNODES / 16, 128, 0, stream>>>(h16, w16t, feat);
  edge_score<128, 5><<<EG, 256, 0, stream>>>(feat, src, dst, attn[2], esc);
  node_aggregate<128, 5, true><<<NG, 256, 0, stream>>>(feat, esc, rowptr, eidx, src, bufA);

  // ---- layer 3: Fout=32 (D=8), no activation ----
  proj(bufA, 128, Wl[3], 32, 128);
  gemm_wmma<128, 32><<<NNODES / 16, 128, 0, stream>>>(h16, w16t, feat);
  edge_score<32, 3><<<EG, 256, 0, stream>>>(feat, src, dst, attn[3], esc);
  node_aggregate<32, 3, false><<<NG, 256, 0, stream>>>(feat, esc, rowptr, eidx, src, bufB);

  int t = NNODES * 8;
  head_mean<<<(t + 255) / 256, 256, 0, stream>>>(bufB, out, NNODES);
}
